// VariableLengthAttention_43413529428725
// MI455X (gfx1250) — matrix operand order repaired and checked by reading, not compiled, not run on hardware
//
#include <hip/hip_runtime.h>

// ---------------------------------------------------------------------------
// VariableLengthAttention for MI455X (gfx1250, wave32, WMMA bf16 f32-acc)
// ---------------------------------------------------------------------------
//  B=16, MEL_LEN=2048, SRC_LEN=512, D_MODEL=D_ENC=256, temperature=16
//  d_out = [output f32 B*512*256] ++ [attn f32 B*512*2048]
//  d_ws  = qb/kb/vT (bf16 intermediates) ++ bf16 copies of mel/text/W*
//  Pipeline:
//    0) streaming f32->bf16 conversion of inputs (pure bandwidth)
//    1) bf16 WMMA projections q,k (row-major) and v^T (transposed store)
//    2) attention: LDS-resident 16x2048 score tile, single-pass softmax,
//       bf16 P copy in LDS -> P@V with raw ds_load_b128 A-fragments,
//       fused LayerNorm
// ---------------------------------------------------------------------------

typedef __attribute__((ext_vector_type(16))) __bf16 v16bf;
typedef __attribute__((ext_vector_type(8)))  float  v8f;

#define BATCH   16
#define MEL_LEN 2048
#define SRC_LEN 512
#define DM      256
#define SLD     2052   // padded LDS f32 score row stride (floats)
#define PLD     2056   // padded LDS bf16 P row stride (halves), 16B-aligned
#define OLD     260    // padded LDS output-tile row stride (floats)

union FragBF {
  v16bf v;
  uint4 u[2];
  unsigned d[8];
  unsigned short h[16];
  __device__ FragBF() {}
};

// Pack two f32 -> two bf16 in one dword (round-half-up + v_perm_b32).
__device__ __forceinline__ unsigned pack_bf16(float a, float b) {
  unsigned ua = __float_as_uint(a) + 0x8000u;
  unsigned ub = __float_as_uint(b) + 0x8000u;
  return __builtin_amdgcn_perm(ub, ua, 0x07060302u);
}
__device__ __forceinline__ unsigned short f2bf1(float a) {
  return (unsigned short)((__float_as_uint(a) + 0x8000u) >> 16);
}

// A-matrix (16x32 bf16) fragment: lanes 0-15 hold K 0..7 (V0-3) and K 16..23
// (V4-7); lanes 16-31 hold K 8..15 / 24..31.  `p` pre-offset (+8 halves for
// upper half-wave); reads p[0..7] and p[16..23].
__device__ __forceinline__ v16bf load_a_bf16(const unsigned short* p) {
  FragBF f;
  f.u[0] = *(const uint4*)(p);
  f.u[1] = *(const uint4*)(p + 16);
  return f.v;
}
// B-matrix (32x16 bf16) fragment: lanes 0-15 hold K 0..7 / 16..23, lanes
// 16-31 hold K 8..15 / 24..31.  `p` pre-offset (+8 halves for upper half-wave).
__device__ __forceinline__ v16bf load_b_bf16(const unsigned short* p) {
  FragBF f;
  f.u[0] = *(const uint4*)(p);
  f.u[1] = *(const uint4*)(p + 16);
  return f.v;
}

__device__ __forceinline__ v8f wmma_bf16(v16bf a, v16bf b, v8f c) {
  return __builtin_amdgcn_wmma_f32_16x16x32_bf16(false, a, false, b,
                                                 (short)0, c, false, false);
}

// ---------------------------------------------------------------------------
// Streaming f32 -> bf16 conversion (one thread = 8 elements).
// ---------------------------------------------------------------------------
__global__ void __launch_bounds__(256)
cvt_bf16_kernel(const float* __restrict__ src, unsigned short* __restrict__ dst,
                int n8) {
  int i = blockIdx.x * 256 + threadIdx.x;
  if (i >= n8) return;
  const float4* s4 = (const float4*)src;
  float4 a = s4[2 * i], b = s4[2 * i + 1];
  uint4 o;
  o.x = pack_bf16(a.x, a.y);
  o.y = pack_bf16(a.z, a.w);
  o.z = pack_bf16(b.x, b.y);
  o.w = pack_bf16(b.z, b.w);
  ((uint4*)dst)[i] = o;
}

// ---------------------------------------------------------------------------
// Projection: Y = X @ W^T  (X: [rows,256] bf16 row-major, W: [256,256] bf16)
// Output bf16; transposed==1 writes per-batch [256][2048] (for V).
// Block = 256 threads (8 waves); 16 rows x 256 cols per block; each wave
// owns two 16x16 N-tiles; inner loop = 6 raw 16B loads + 2 WMMAs.
// ---------------------------------------------------------------------------
__global__ void __launch_bounds__(256)
proj16_kernel(const unsigned short* __restrict__ X,
              const unsigned short* __restrict__ W,
              unsigned short* __restrict__ Y, int transposed) {
  const int tid   = threadIdx.x;
  const int lane  = tid & 31;
  const int wv    = tid >> 5;
  const int lr    = lane & 15;
  const int aoff  = (lane & 16) ? 8 : 0;
  const int boff  = (lane & 16) ? 8 : 0;
  const int mbase = (lane & 16) ? 8 : 0;
  const int grow0 = blockIdx.x << 4;

  const unsigned short* xrow  = X + (size_t)(grow0 + lr) * DM + aoff;
  const int             n0    = wv * 2;
  const unsigned short* wrow0 = W + (size_t)(n0 * 16 + lr) * DM + boff;
  const unsigned short* wrow1 = W + (size_t)((n0 + 1) * 16 + lr) * DM + boff;

  v8f acc0 = {}; v8f acc1 = {};
#pragma unroll
  for (int ks = 0; ks < 8; ++ks) {
    v16bf a  = load_a_bf16(xrow + ks * 32);
    v16bf b0 = load_b_bf16(wrow0 + ks * 32);
    v16bf b1 = load_b_bf16(wrow1 + ks * 32);
    acc0 = wmma_bf16(a, b0, acc0);
    acc1 = wmma_bf16(a, b1, acc1);
  }

  if (!transposed) {
#pragma unroll
    for (int j = 0; j < 8; ++j) {
      Y[(size_t)(grow0 + mbase + j) * DM + n0 * 16 + lr]       = f2bf1(acc0[j]);
      Y[(size_t)(grow0 + mbase + j) * DM + (n0 + 1) * 16 + lr] = f2bf1(acc1[j]);
    }
  } else {
    // vT[b][n][s]: C-fragment rows (M = s) are contiguous -> one uint4 store.
    const int b  = grow0 >> 11;               // 2048 mel rows per batch
    const int s0 = (grow0 & (MEL_LEN - 1)) + mbase;
    FragBF h0, h1;
#pragma unroll
    for (int j = 0; j < 4; ++j) {
      h0.d[j] = pack_bf16(acc0[2 * j], acc0[2 * j + 1]);
      h1.d[j] = pack_bf16(acc1[2 * j], acc1[2 * j + 1]);
    }
    unsigned short* yb = Y + (size_t)b * DM * MEL_LEN;
    *(uint4*)(yb + (size_t)(n0 * 16 + lr) * MEL_LEN + s0)       = h0.u[0];
    *(uint4*)(yb + (size_t)((n0 + 1) * 16 + lr) * MEL_LEN + s0) = h1.u[0];
  }
}

// ---------------------------------------------------------------------------
// Attention: one block per (batch, 16-query tile).  LDS layout:
//   S  : 16 x SLD f32 scores (O-tile aliases this region after attn is out)
//   Pb : 16 x PLD bf16 probabilities (A-fragments for P@V, raw ds_load_b128)
//   red/lnb : reductions
// ---------------------------------------------------------------------------
__global__ void __launch_bounds__(256)
attn16_kernel(const unsigned short* __restrict__ qb,
              const unsigned short* __restrict__ kb,
              const unsigned short* __restrict__ vtb,
              const unsigned char* __restrict__ mel_mask,
              const unsigned char* __restrict__ src_mask,
              const float* __restrict__ gamma,
              const float* __restrict__ beta,
              float* __restrict__ out,     // [B,512,256]
              float* __restrict__ attn) {  // [B,512,2048]
  extern __shared__ char smemraw[];
  float*          S   = (float*)smemraw;                 // 16*SLD f32
  float*          O   = S;                               // alias (16*OLD f32)
  unsigned short* Pb  = (unsigned short*)(S + 16 * SLD); // 16*PLD bf16
  float*          red = (float*)(Pb + 16 * PLD);         // 512
  float*          lnb = red + 512;                       // 32

  const int tid   = threadIdx.x;
  const int lane  = tid & 31;
  const int wv    = tid >> 5;
  const int lr    = lane & 15;
  const int aoff  = (lane & 16) ? 8 : 0;
  const int boff  = (lane & 16) ? 8 : 0;
  const int mbase = (lane & 16) ? 8 : 0;

  const int b     = blockIdx.x >> 5;
  const int trow0 = (blockIdx.x & 31) << 4;

  // ---- Phase 1: S = Q K^T, waves split the 128 key tiles ----------------
  const unsigned short* qrow =
      qb + (size_t)(b * SRC_LEN + trow0 + lr) * DM + aoff;
  v16bf qf[8];
#pragma unroll
  for (int ks = 0; ks < 8; ++ks) qf[ks] = load_a_bf16(qrow + ks * 32);

  const unsigned short* kbb = kb + (size_t)b * MEL_LEN * DM;
  for (int i = 0; i < 16; ++i) {
    const int kt = wv + i * 8;
    const unsigned short* krow = kbb + (size_t)(kt * 16 + lr) * DM + boff;
    v8f acc = {};
#pragma unroll
    for (int ks = 0; ks < 8; ++ks)
      acc = wmma_bf16(qf[ks], load_b_bf16(krow + ks * 32), acc);
#pragma unroll
    for (int j = 0; j < 8; ++j)
      S[(mbase + j) * SLD + kt * 16 + lr] = acc[j];
  }
  __syncthreads();

  // ---- Phase 2: scale, mel-mask, softmax, src-mask, bf16 P copy ---------
  const int r   = tid >> 4;                    // query row 0..15
  const int seg = tid & 15;                    // 128-wide column segment
  const int c0  = seg * 128;
  const float NEGINF = -__builtin_huge_valf();

  float4* Sr4 = (float4*)(S + r * SLD + c0);
  const uchar4* mm4 = (const uchar4*)(mel_mask + (size_t)b * MEL_LEN + c0);

  float mx = -3.4e38f;
#pragma unroll 4
  for (int i = 0; i < 32; ++i) {
    float4 s = Sr4[i];
    uchar4 m = mm4[i];
    s.x = m.x ? NEGINF : s.x * 0.0625f;
    s.y = m.y ? NEGINF : s.y * 0.0625f;
    s.z = m.z ? NEGINF : s.z * 0.0625f;
    s.w = m.w ? NEGINF : s.w * 0.0625f;
    Sr4[i] = s;
    mx = fmaxf(mx, fmaxf(fmaxf(s.x, s.y), fmaxf(s.z, s.w)));
  }
  red[tid] = mx;
  __syncthreads();
  float rmx = -3.4e38f;
#pragma unroll
  for (int i = 0; i < 16; ++i) rmx = fmaxf(rmx, red[(r << 4) + i]);
  __syncthreads();

  float sum = 0.f;
#pragma unroll 4
  for (int i = 0; i < 32; ++i) {
    float4 s = Sr4[i];
    s.x = __expf(s.x - rmx);
    s.y = __expf(s.y - rmx);
    s.z = __expf(s.z - rmx);
    s.w = __expf(s.w - rmx);
    Sr4[i] = s;
    sum += (s.x + s.y) + (s.z + s.w);
  }
  red[tid] = sum;
  __syncthreads();
  float tot = 0.f;
#pragma unroll
  for (int i = 0; i < 16; ++i) tot += red[(r << 4) + i];
  float scale = (tot > 0.f) ? (1.f / tot) : 0.f;
  if (src_mask[(size_t)b * SRC_LEN + trow0 + r]) scale = 0.f;

  unsigned* Pr32 = (unsigned*)(Pb + r * PLD + c0);
#pragma unroll 4
  for (int i = 0; i < 32; ++i) {
    float4 s = Sr4[i];
    s.x *= scale; s.y *= scale; s.z *= scale; s.w *= scale;
    Sr4[i] = s;
    Pr32[2 * i]     = pack_bf16(s.x, s.y);
    Pr32[2 * i + 1] = pack_bf16(s.z, s.w);
  }
  __syncthreads();

  // coalesced attn write-out (16*2048 f32, linear over the tile)
  float* ao = attn + ((size_t)b * SRC_LEN + trow0) * MEL_LEN;
  for (int it = 0; it < 32; ++it) {
    int idx = it * 1024 + tid * 4;
    int rr = idx >> 11;
    int cc = idx & (MEL_LEN - 1);
    *(float4*)(ao + idx) = *(const float4*)&S[rr * SLD + cc];
  }
  __syncthreads();   // S is dead now; O-tile aliases it below

  // ---- Phase 3: O = P V, waves split the 16 N-tiles (2 each) ------------
  const unsigned short* vb = vtb + (size_t)b * DM * MEL_LEN;
  const int n0 = wv * 32;
  const unsigned short* v0 = vb + (size_t)(n0 + lr) * MEL_LEN + boff;
  const unsigned short* v1 = vb + (size_t)(n0 + 16 + lr) * MEL_LEN + boff;
  const unsigned short* prow = Pb + lr * PLD + aoff;

  v8f o0 = {}; v8f o1 = {};
  for (int st = 0; st < 64; ++st) {
    v16bf a = load_a_bf16(prow + st * 32);     // raw ds_load_b128 x2, no cvt
    o0 = wmma_bf16(a, load_b_bf16(v0 + st * 32), o0);
    o1 = wmma_bf16(a, load_b_bf16(v1 + st * 32), o1);
  }
#pragma unroll
  for (int j = 0; j < 8; ++j) {
    O[(mbase + j) * OLD + n0 + lr]      = o0[j];
    O[(mbase + j) * OLD + n0 + 16 + lr] = o1[j];
  }
  __syncthreads();

  // ---- Phase 4: LayerNorm over d_model=256 ------------------------------
  float s1 = 0.f, s2 = 0.f;
  const int nb = seg * 16;
#pragma unroll
  for (int i = 0; i < 16; ++i) {
    float v = O[r * OLD + nb + i];
    s1 += v; s2 += v * v;
  }
  red[tid] = s1; red[256 + tid] = s2;
  __syncthreads();
  if (tid < 16) {
    float a = 0.f, q = 0.f;
#pragma unroll
    for (int i = 0; i < 16; ++i) {
      a += red[(tid << 4) + i];
      q += red[256 + (tid << 4) + i];
    }
    float mu  = a * (1.f / 256.f);
    float var = q * (1.f / 256.f) - mu * mu;
    lnb[tid]      = mu;
    lnb[16 + tid] = rsqrtf(var + 1e-5f);
  }
  __syncthreads();

  float* op = out + ((size_t)b * SRC_LEN + trow0) * DM;
#pragma unroll
  for (int it = 0; it < 16; ++it) {
    int idx = it * 256 + tid;
    int rr = idx >> 8;
    int nn = idx & (DM - 1);
    op[idx] = (O[rr * OLD + nn] - lnb[rr]) * lnb[16 + rr] * gamma[nn] + beta[nn];
  }
}

// ---------------------------------------------------------------------------
extern "C" void kernel_launch(void* const* d_in, const int* in_sizes, int n_in,
                              void* d_out, int out_size, void* d_ws, size_t ws_size,
                              hipStream_t stream) {
  (void)in_sizes; (void)n_in; (void)out_size; (void)ws_size;

  const float*         mel      = (const float*)d_in[0];
  const float*         text     = (const float*)d_in[1];
  const unsigned char* mel_mask = (const unsigned char*)d_in[2];
  const unsigned char* src_mask = (const unsigned char*)d_in[3];
  const float*         Wq       = (const float*)d_in[4];
  const float*         Wk       = (const float*)d_in[5];
  const float*         Wv       = (const float*)d_in[6];
  const float*         gamma    = (const float*)d_in[7];
  const float*         beta     = (const float*)d_in[8];

  float* out  = (float*)d_out;                              // B*512*256
  float* attn = out + (size_t)BATCH * SRC_LEN * DM;         // B*512*2048

  const size_t nQ = (size_t)BATCH * SRC_LEN * DM;           // 2,097,152
  const size_t nK = (size_t)BATCH * MEL_LEN * DM;           // 8,388,608
  const size_t nW = (size_t)DM * DM;                        // 65,536

  unsigned short* qb    = (unsigned short*)d_ws;            // bf16 q
  unsigned short* kb    = qb    + nQ;                       // bf16 k
  unsigned short* vtb   = kb    + nK;                       // bf16 v^T
  unsigned short* melb  = vtb   + nK;                       // bf16 mel
  unsigned short* textb = melb  + nK;                       // bf16 text
  unsigned short* wqb   = textb + nQ;                       // bf16 Wq
  unsigned short* wkb   = wqb   + nW;                       // bf16 Wk
  unsigned short* wvb   = wkb   + nW;                       // bf16 Wv

  dim3 blk(256);
  // 0) input conversion (streaming, bandwidth-bound)
  cvt_bf16_kernel<<<dim3((unsigned)(nK / 8 / 256)), blk, 0, stream>>>(mel,  melb,  (int)(nK / 8));
  cvt_bf16_kernel<<<dim3((unsigned)(nQ / 8 / 256)), blk, 0, stream>>>(text, textb, (int)(nQ / 8));
  cvt_bf16_kernel<<<dim3((unsigned)(nW / 8 / 256)), blk, 0, stream>>>(Wq, wqb, (int)(nW / 8));
  cvt_bf16_kernel<<<dim3((unsigned)(nW / 8 / 256)), blk, 0, stream>>>(Wk, wkb, (int)(nW / 8));
  cvt_bf16_kernel<<<dim3((unsigned)(nW / 8 / 256)), blk, 0, stream>>>(Wv, wvb, (int)(nW / 8));

  // 1) projections (pure bf16 WMMA)
  proj16_kernel<<<dim3(BATCH * SRC_LEN / 16), blk, 0, stream>>>(textb, wqb, qb, 0);
  proj16_kernel<<<dim3(BATCH * MEL_LEN / 16), blk, 0, stream>>>(melb,  wkb, kb, 0);
  proj16_kernel<<<dim3(BATCH * MEL_LEN / 16), blk, 0, stream>>>(melb,  wvb, vtb, 1);

  // 2) attention + softmax + LayerNorm
  const size_t sh = (size_t)(16 * SLD) * sizeof(float)          // S (+O alias)
                  + (size_t)(16 * PLD) * sizeof(unsigned short) // Pb
                  + (size_t)(512 + 32) * sizeof(float);         // red + lnb
  (void)hipFuncSetAttribute((const void*)attn16_kernel,
                            hipFuncAttributeMaxDynamicSharedMemorySize, (int)sh);
  attn16_kernel<<<dim3(BATCH * (SRC_LEN / 16)), blk, sh, stream>>>(
      qb, kb, vtb, mel_mask, src_mask, gamma, beta, out, attn);
}
